// MultiheadAttention_69140383531329
// MI455X (gfx1250) — compile-verified, matrix operand order checked by reading, not run on hardware
//
#include <hip/hip_runtime.h>
#include <hip/hip_bf16.h>

// ---------------------------------------------------------------------------
// MultiheadAttention for MI455X (gfx1250): all GEMM work via
// v_wmma_f32_16x16x32_f16 (f16 inputs, f32 accumulate).
//  - Projection/output GEMMs: W k-slice staged once per block into LDS (f16),
//    32x64 C strip per wave (8 WMMAs per k-step), A prefetched.
//  - Flash attention: K blocks staged via async global->LDS copies
//    (ASYNCcnt path), V transposed into LDS, online softmax with lane
//    reductions, P relayout through LDS, all matmuls on WMMA.
// ---------------------------------------------------------------------------

#define EMBED   1024
#define NHEADS  16
#define HDIM    64
#define BATCH   2
#define SEQ     2048
#define MROWS   (BATCH * SEQ)   // 4096 flattened (b, s) rows

typedef __attribute__((ext_vector_type(16))) _Float16 v16h;
typedef __attribute__((ext_vector_type(8)))  _Float16 v8h;
typedef __attribute__((ext_vector_type(8)))  float    v8f;

union F16x16 { v16h v; v8h h[2]; _Float16 e[16]; };
union F16x8  { v8h  v; _Float16 e[8]; };

// Load 8 consecutive f16 (one half of an A/B fragment chunk pair).
__device__ inline v8h ld8(const _Float16* p) {
    return *reinterpret_cast<const v8h*>(p);
}
// Same, but convert from fp32 source on the fly (two 128-bit loads + cvt).
__device__ inline v8h ld8(const float* p) {
    const float4* q = reinterpret_cast<const float4*>(p);
    float4 a = q[0], b = q[1];
    F16x8 r;
    r.e[0] = (_Float16)a.x; r.e[1] = (_Float16)a.y;
    r.e[2] = (_Float16)a.z; r.e[3] = (_Float16)a.w;
    r.e[4] = (_Float16)b.x; r.e[5] = (_Float16)b.y;
    r.e[6] = (_Float16)b.z; r.e[7] = (_Float16)b.w;
    return r.v;
}

__device__ inline v8f wmma_f16(const F16x16& a, const F16x16& b, v8f c) {
    return __builtin_amdgcn_wmma_f32_16x16x32_f16(
        false, a.v, false, b.v, (short)0, c, false, false);
}

// Async 16B global -> LDS copy (VFLAT GLOBAL_LOAD_ASYNC_TO_LDS_B128,
// tracked with ASYNCcnt).  lds_off is the 32-bit LDS byte offset.
__device__ inline void async_copy_b128(unsigned lds_off, const void* gaddr) {
    asm volatile("global_load_async_to_lds_b128 %0, %1, off"
                 :: "v"(lds_off), "v"((unsigned long long)gaddr)
                 : "memory");
}
__device__ inline void wait_asynccnt0() {
#if defined(__has_builtin) && __has_builtin(__builtin_amdgcn_s_wait_asynccnt)
    __builtin_amdgcn_s_wait_asynccnt(0);
#else
    asm volatile("s_wait_asynccnt 0x0" ::: "memory");
#endif
}

// ---------------------------------------------------------------------------
// GEMM: C[M,N] = A[M,K] @ W[N,K]^T.  256 threads = 8 waves.
// Block tile: 256 rows x 64 cols.  Wave tile: 32 x 64 (2 M-frags x 4 N-frags
// -> 8 WMMAs per 32-wide k-step).  W k-slice (64x32) converted fp32->f16 once
// per block into LDS; all waves read B-fragments from LDS.
// AT = float (convert on the fly) or _Float16 (direct).  F16OUT: C dtype.
// ---------------------------------------------------------------------------
template <typename AT, bool F16OUT>
__global__ __launch_bounds__(256) void gemm_xwT_kernel(
    const AT* __restrict__ A, const float* __restrict__ W,
    void* __restrict__ Cout, int M, int N, int K)
{
    __shared__ _Float16 sW[64][32];      // [n][k] rows 64B (16B-aligned)

    const int tid  = threadIdx.x;
    const int wave = tid >> 5;
    const int lane = tid & 31;
    const int lm   = lane & 15;          // M (or N) index within tile
    const int lk   = (lane >> 4) * 8;    // K sub-chunk base per lane half

    const int m0 = (blockIdx.x * 8 + wave) * 32;
    const int n0 = blockIdx.y * 64;

    // Cooperative W staging map: thread -> (row n, 8-wide k chunk).
    const int wn = tid >> 2;             // 0..63
    const int wc = (tid & 3) * 8;        // 0,8,16,24
    const float* wsrc = W + (size_t)(n0 + wn) * K + wc;

    v8f acc[2][4] = {};

    const AT* arow0 = A + (size_t)(m0 + lm) * K;
    const AT* arow1 = A + (size_t)(m0 + 16 + lm) * K;

    for (int k0 = 0; k0 < K; k0 += 32) {
        __syncthreads();                          // previous k-step done with sW
        *(v8h*)&sW[wn][wc] = ld8(wsrc + k0);      // fp32->f16 convert, once/block
        __syncthreads();

        // Prefetch next k-slice of the A stream (global_prefetch_b8).
        __builtin_prefetch(arow0 + k0 + 32, 0, 3);
        __builtin_prefetch(arow1 + k0 + 32, 0, 3);

        F16x16 a0, a1;
        a0.h[0] = ld8(arow0 + k0 + lk);
        a0.h[1] = ld8(arow0 + k0 + lk + 16);
        a1.h[0] = ld8(arow1 + k0 + lk);
        a1.h[1] = ld8(arow1 + k0 + lk + 16);

        #pragma unroll
        for (int t = 0; t < 4; ++t) {
            F16x16 bf;
            bf.h[0] = *(const v8h*)&sW[t * 16 + lm][lk];
            bf.h[1] = *(const v8h*)&sW[t * 16 + lm][lk + 16];
            acc[0][t] = wmma_f16(a0, bf, acc[0][t]);
            acc[1][t] = wmma_f16(a1, bf, acc[1][t]);
        }
    }

    // C/D layout: lane half selects row group of 8; lane&15 selects column.
    #pragma unroll
    for (int half = 0; half < 2; ++half) {
        const int row0 = m0 + half * 16 + (lane >> 4) * 8;
        #pragma unroll
        for (int t = 0; t < 4; ++t) {
            const int col = n0 + t * 16 + lm;
            #pragma unroll
            for (int r = 0; r < 8; ++r) {
                if constexpr (F16OUT)
                    ((_Float16*)Cout)[(size_t)(row0 + r) * N + col] =
                        (_Float16)acc[half][t][r];
                else
                    ((float*)Cout)[(size_t)(row0 + r) * N + col] = acc[half][t][r];
            }
        }
    }
}

// ---------------------------------------------------------------------------
// Flash attention: grid (B*H, SEQ/64), 128 threads = 4 waves; wave w handles
// 16 query rows.  Key blocks of 32: K staged by async global->LDS copies,
// V transposed into LDS; online softmax with 16-lane shfl_xor reductions.
// ---------------------------------------------------------------------------
__global__ __launch_bounds__(128) void attn_kernel(
    const _Float16* __restrict__ Qf, const _Float16* __restrict__ Kf,
    const _Float16* __restrict__ Vf, _Float16* __restrict__ AO)
{
    __shared__ _Float16 sK [32][72];      // [key][d]   rows 144B (16B-aligned)
    __shared__ _Float16 sVT[64][48];      // [d][key]   rows  96B (16B-aligned)
    __shared__ _Float16 sP [4][16][32];   // per-wave P tile, C->A relayout

    const int tid  = threadIdx.x;
    const int wave = tid >> 5;
    const int lane = tid & 31;
    const int lm   = lane & 15;
    const int lk   = (lane >> 4) * 8;

    const int b = blockIdx.x >> 4;
    const int h = blockIdx.x & 15;
    const int q0 = blockIdx.y * 64 + wave * 16;

    // Q A-fragments for the two head-dim K-steps (d 0..31, 32..63).
    F16x16 qa[2];
    {
        const _Float16* qrow = Qf + ((size_t)b * SEQ + q0 + lm) * EMBED + h * HDIM;
        #pragma unroll
        for (int s = 0; s < 2; ++s) {
            qa[s].h[0] = ld8(qrow + s * 32 + lk);
            qa[s].h[1] = ld8(qrow + s * 32 + lk + 16);
        }
    }

    v8f o[4] = {};
    float mrow[8], lrow[8];
    #pragma unroll
    for (int r = 0; r < 8; ++r) { mrow[r] = -1e30f; lrow[r] = 0.0f; }

    // Per-thread staging map: 16B chunk of one key row.
    const int chunk = tid & 7;      // d chunk (8 f16)
    const int rowh  = tid >> 3;     // 0..15

    for (int kb0 = 0; kb0 < SEQ; kb0 += 32) {
        __syncthreads();   // previous iteration done reading sK/sVT
        #pragma unroll
        for (int half = 0; half < 2; ++half) {
            const int key = rowh + half * 16;
            const size_t goff = ((size_t)b * SEQ + kb0 + key) * EMBED + h * HDIM + chunk * 8;
            // K tile: async copy straight into LDS (no VGPR round trip).
            async_copy_b128((unsigned)(uintptr_t)&sK[key][chunk * 8], Kf + goff);
            // V tile: load + transpose into LDS.
            F16x8 vv; vv.v = ld8(Vf + goff);
            #pragma unroll
            for (int j = 0; j < 8; ++j) sVT[chunk * 8 + j][key] = vv.e[j];
        }
        wait_asynccnt0();          // async K copies LDS-visible
        __syncthreads();

        // S = Q @ K^T  (two 16-key tiles, two d-steps each)
        v8f sc[2] = {};
        #pragma unroll
        for (int t = 0; t < 2; ++t) {
            const int key = t * 16 + lm;
            #pragma unroll
            for (int s = 0; s < 2; ++s) {
                F16x16 kbf;
                kbf.h[0] = *(const v8h*)&sK[key][s * 32 + lk];
                kbf.h[1] = *(const v8h*)&sK[key][s * 32 + lk + 16];
                sc[t] = wmma_f16(qa[s], kbf, sc[t]);
            }
        }

        // Online softmax.  Row r lives in VGPR slot r across 16 lanes of one
        // half-wave; xor masks 1..8 stay inside the half.
        float pscale[8];
        #pragma unroll
        for (int r = 0; r < 8; ++r) {
            float s0 = sc[0][r] * 0.125f;   // 1/sqrt(HDIM)
            float s1 = sc[1][r] * 0.125f;
            sc[0][r] = s0; sc[1][r] = s1;
            float mx = fmaxf(s0, s1);
            #pragma unroll
            for (int d = 1; d < 16; d <<= 1) mx = fmaxf(mx, __shfl_xor(mx, d, 32));
            const float nm = fmaxf(mrow[r], mx);
            pscale[r] = __expf(mrow[r] - nm);
            mrow[r] = nm;
        }
        #pragma unroll
        for (int r = 0; r < 8; ++r) {
            float p0 = __expf(sc[0][r] - mrow[r]);
            float p1 = __expf(sc[1][r] - mrow[r]);
            sc[0][r] = p0; sc[1][r] = p1;
            float sum = p0 + p1;
            #pragma unroll
            for (int d = 1; d < 16; d <<= 1) sum += __shfl_xor(sum, d, 32);
            lrow[r] = lrow[r] * pscale[r] + sum;
        }
        #pragma unroll
        for (int t = 0; t < 4; ++t)
            #pragma unroll
            for (int r = 0; r < 8; ++r) o[t][r] *= pscale[r];

        // Relayout P (C-layout) -> A-layout through this wave's LDS slab.
        {
            const int rb = (lane >> 4) * 8;
            #pragma unroll
            for (int t = 0; t < 2; ++t)
                #pragma unroll
                for (int r = 0; r < 8; ++r)
                    sP[wave][rb + r][t * 16 + lm] = (_Float16)sc[t][r];
        }
        F16x16 pa;   // LDS in-order per wave; compiler inserts s_wait_dscnt
        pa.h[0] = *(const v8h*)&sP[wave][lm][lk];
        pa.h[1] = *(const v8h*)&sP[wave][lm][lk + 16];

        // O += P @ V  (K-dim = 32 keys, 4 d-tiles)
        #pragma unroll
        for (int t = 0; t < 4; ++t) {
            const int dcol = t * 16 + lm;
            F16x16 vbf;
            vbf.h[0] = *(const v8h*)&sVT[dcol][lk];
            vbf.h[1] = *(const v8h*)&sVT[dcol][lk + 16];
            o[t] = wmma_f16(pa, vbf, o[t]);
        }
    }

    // Normalize and emit f16 attention output [B,S,E].
    const int rb = (lane >> 4) * 8;
    #pragma unroll
    for (int r = 0; r < 8; ++r) {
        const float inv = 1.0f / lrow[r];
        const size_t rowoff = ((size_t)b * SEQ + q0 + rb + r) * EMBED + h * HDIM;
        #pragma unroll
        for (int t = 0; t < 4; ++t)
            AO[rowoff + t * 16 + lm] = (_Float16)(o[t][r] * inv);
    }
}

// ---------------------------------------------------------------------------
extern "C" void kernel_launch(void* const* d_in, const int* in_sizes, int n_in,
                              void* d_out, int out_size, void* d_ws, size_t ws_size,
                              hipStream_t stream) {
    (void)in_sizes; (void)n_in; (void)out_size; (void)ws_size;
    const float* q  = (const float*)d_in[0];
    const float* k  = (const float*)d_in[1];
    const float* v  = (const float*)d_in[2];
    const float* Wq = (const float*)d_in[3];
    const float* Wk = (const float*)d_in[4];
    const float* Wv = (const float*)d_in[5];
    const float* Wo = (const float*)d_in[6];

    // Workspace: 4 x (4096*1024) f16 = 32 MB (fits in the 192 MB L2).
    const size_t PLANE = (size_t)MROWS * EMBED;
    _Float16* Qf = (_Float16*)d_ws;
    _Float16* Kf = Qf + PLANE;
    _Float16* Vf = Kf + PLANE;
    _Float16* AO = Vf + PLANE;

    dim3 gblk(MROWS / 256, EMBED / 64);   // 16 x 16 blocks, 8 waves each
    gemm_xwT_kernel<float, true><<<gblk, 256, 0, stream>>>(q, Wq, Qf, MROWS, EMBED, EMBED);
    gemm_xwT_kernel<float, true><<<gblk, 256, 0, stream>>>(k, Wk, Kf, MROWS, EMBED, EMBED);
    gemm_xwT_kernel<float, true><<<gblk, 256, 0, stream>>>(v, Wv, Vf, MROWS, EMBED, EMBED);

    attn_kernel<<<dim3(BATCH * NHEADS, SEQ / 64), 128, 0, stream>>>(Qf, Kf, Vf, AO);

    gemm_xwT_kernel<_Float16, false><<<gblk, 256, 0, stream>>>(AO, Wo, (float*)d_out,
                                                               MROWS, EMBED, EMBED);
}
